// StandardBlock_19610820673717
// MI455X (gfx1250) — matrix operand order repaired and checked by reading, not compiled, not run on hardware
//
#include <hip/hip_runtime.h>
#include <math.h>

// ---------------- problem constants (from reference) ----------------
#define T_TOK 8192          // B*S
#define DD    1024          // model dim
#define EE    8             // experts
#define KSEL  2             // top-k
#define HH    4096          // expert hidden
#define PMAX  (T_TOK*KSEL + EE*128)   // 17408: max padded gathered rows

typedef __bf16 v16bf __attribute__((ext_vector_type(16)));
typedef float  v8f   __attribute__((ext_vector_type(8)));

__device__ __forceinline__ unsigned short f32_to_bf16(float f) {
  unsigned u = __float_as_uint(f);
  unsigned r = u + 0x7FFFu + ((u >> 16) & 1u);   // round-to-nearest-even
  return (unsigned short)(r >> 16);
}
__device__ __forceinline__ float bf16_to_f32(unsigned short s) {
  return __uint_as_float(((unsigned)s) << 16);
}
__device__ __forceinline__ float gelu_exact(float x) {
  return 0.5f * x * (1.0f + erff(x * 0.70710678118654752f));
}

// ---------------- f32 -> bf16 conversion ----------------
__global__ void cvt_bf16_kernel(const float* __restrict__ src,
                                unsigned short* __restrict__ dst, long n) {
  long i = (long)blockIdx.x * blockDim.x + threadIdx.x;
  if (i < n) dst[i] = f32_to_bf16(src[i]);
}

__global__ void fill_int_kernel(int* __restrict__ p, int val, int n) {
  int i = blockIdx.x * blockDim.x + threadIdx.x;
  if (i < n) p[i] = val;
}

// ---------------- tiled bf16 WMMA GEMM (double-buffered) ----------------
// C[M,N] = act( A[M,Kd] @ B[Kd,N] + bias ),  A/B bf16 row-major, acc f32.
// Block: 128x128, 256 threads = 8 waves; wave w owns rows [16w,16w+16) x all
// 8 column tiles.  BK = 32 (one v_wmma_f32_16x16x32_bf16 K-step per tile).
// Ping-pong LDS buffers: global loads for step i+1 overlap WMMAs of step i;
// one workgroup barrier per K-step.
// Optional row gather (rowidx, -1 => zero row) and per-block expert select
// via 128-aligned segment offsets (seg_off[EE+1]).

__device__ __forceinline__ void g_load_tiles(
    const unsigned short* __restrict__ A, long arow, int Kd,
    const unsigned short* __restrict__ B, int N, int colbase,
    int k0, int ahalf, int bk, int bc,
    uint4& a0, uint4& a1, uint4& b0, uint4& b1)
{
  if (arow >= 0) {
    const uint4* src = (const uint4*)(A + arow * (long)Kd + k0 + ahalf * 16);
    a0 = src[0]; a1 = src[1];
  } else {
    a0 = make_uint4(0u, 0u, 0u, 0u);
    a1 = make_uint4(0u, 0u, 0u, 0u);
  }
  const uint4* bsrc = (const uint4*)(B + (long)(k0 + bk) * N + colbase + bc);
  b0 = bsrc[0]; b1 = bsrc[1];
}

__device__ __forceinline__ void lds_store_tiles(
    unsigned short* __restrict__ sA, unsigned short* __restrict__ sB,
    int ar, int ahalf, int bk, int bc,
    uint4 a0, uint4 a1, uint4 b0, uint4 b1)
{
  uint4* dst = (uint4*)(sA + ar * 48 + ahalf * 16);
  dst[0] = a0; dst[1] = a1;
  unsigned v[8] = {b0.x, b0.y, b0.z, b0.w, b1.x, b1.y, b1.z, b1.w};
  #pragma unroll
  for (int j = 0; j < 8; ++j) {   // transpose to sB[col][k]
    sB[(bc + 2*j    ) * 48 + bk] = (unsigned short)(v[j] & 0xFFFFu);
    sB[(bc + 2*j + 1) * 48 + bk] = (unsigned short)(v[j] >> 16);
  }
}

// B fragment (32x16 bf16) from transposed LDS tile
__device__ __forceinline__ v16bf load_bfrag(const unsigned short* __restrict__ cB,
                                            int c, int ln, int kh) {
  union { v16bf v; unsigned u[8]; } bfm;
  #pragma unroll
  for (int v = 0; v < 8; ++v) {
    int kk = (v < 4 ? 2 * v : 16 + 2 * (v - 4)) + 8 * kh;
    bfm.u[v] = *(const unsigned*)(cB + (c * 16 + ln) * 48 + kk);
  }
  return bfm.v;
}

template<bool GELU, bool OUT_BF16>
__global__ __launch_bounds__(256)
void gemm_bf16_wmma(const unsigned short* __restrict__ A,
                    const int* __restrict__ rowidx,
                    const unsigned short* __restrict__ B,
                    const int* __restrict__ seg_off,
                    long bStrideExpert,
                    const float* __restrict__ bias,
                    long biasStrideExpert,
                    void* __restrict__ Cout,
                    int N, int Kd)
{
  // ping-pong LDS tiles, row stride 48 elems (96 B: 16B-aligned subrows, pad)
  __shared__ unsigned short sA[2][128 * 48];   // [buf][row][k]
  __shared__ unsigned short sB[2][128 * 48];   // [buf][col][k] (transposed)

  const int tid  = threadIdx.x;
  const int lane = tid & 31;
  const int wv   = tid >> 5;       // wave 0..7
  const int kh   = lane >> 4;      // lane half
  const int ln   = lane & 15;
  const int rowbase = blockIdx.y * 128;
  const int colbase = blockIdx.x * 128;

  if (seg_off) {                   // expert-uniform block (offsets 128-aligned)
    int e = 0;
    while (e < EE - 1 && rowbase >= seg_off[e + 1]) ++e;
    B    += (long)e * bStrideExpert;
    bias += (long)e * biasStrideExpert;
  }

  v8f acc[8];
  #pragma unroll
  for (int c = 0; c < 8; ++c)
    #pragma unroll
    for (int i = 0; i < 8; ++i) acc[c][i] = 0.0f;

  // cooperative-load roles
  const int ar    = tid >> 1;        // A row 0..127
  const int ahalf = tid & 1;         // 16-element half of the 32-wide A subrow
  long arow;
  {
    int gr = rowbase + ar;
    if (rowidx) gr = rowidx[gr];     // may be -1 (padding row)
    arow = gr;
  }
  const int bk = tid >> 3;           // B k-row 0..31
  const int bc = (tid & 7) * 16;     // B col chunk base 0..112

  // prologue: stage tile for k0 = 0 into buffer 0
  uint4 a0, a1, b0, b1;
  g_load_tiles(A, arow, Kd, B, N, colbase, 0, ahalf, bk, bc, a0, a1, b0, b1);
  lds_store_tiles(sA[0], sB[0], ar, ahalf, bk, bc, a0, a1, b0, b1);
  __syncthreads();

  int cur = 0;
  for (int k0 = 0; k0 < Kd; k0 += 32) {
    const bool have_next = (k0 + 32) < Kd;
    if (have_next) {   // issue next tile's global loads before computing
      g_load_tiles(A, arow, Kd, B, N, colbase, k0 + 32, ahalf, bk, bc,
                   a0, a1, b0, b1);
    }

    // ---- A fragment (16x32 bf16): ISA layout, VGPR v holds K pair ----
    const unsigned short* cA = sA[cur];
    const unsigned short* cB = sB[cur];
    union { v16bf v; unsigned u[8]; } af;
    #pragma unroll
    for (int v = 0; v < 8; ++v) {
      int kk = (v < 4 ? 2*v : 16 + 2*(v - 4)) + 8 * kh;
      af.u[v] = *(const unsigned*)(cA + (wv * 16 + ln) * 48 + kk);
    }
    // ---- 8 column tiles; A-operand reuse hint on back-to-back WMMAs.
    // reuse flags must be literal constants: peel tile 0 (no reuse), then
    // tiles 1..7 with reuse_a=true (identical back-to-back instructions).
    {
      v16bf bf0 = load_bfrag(cB, 0, ln, kh);
      acc[0] = __builtin_amdgcn_wmma_f32_16x16x32_bf16(
          false, af.v, false, bf0, (short)0, acc[0], false, false);
    }
    #pragma unroll
    for (int c = 1; c < 8; ++c) {
      v16bf bfc = load_bfrag(cB, c, ln, kh);
      acc[c] = __builtin_amdgcn_wmma_f32_16x16x32_bf16(
          false, af.v, false, bfc, (short)0, acc[c], true, false);
    }

    if (have_next) {
      lds_store_tiles(sA[cur ^ 1], sB[cur ^ 1], ar, ahalf, bk, bc,
                      a0, a1, b0, b1);
      __syncthreads();
      cur ^= 1;
    }
  }

  // ---- epilogue: bias (+ GELU), store ----
  #pragma unroll
  for (int c = 0; c < 8; ++c) {
    int ng = colbase + c * 16 + ln;
    float bv = bias[ng];
    #pragma unroll
    for (int rr = 0; rr < 8; ++rr) {
      int mg = rowbase + wv * 16 + rr + 8 * kh;   // C layout: VGPR r -> M=r / 8+r
      float val = acc[c][rr] + bv;
      if (GELU) val = gelu_exact(val);
      if (OUT_BF16)
        ((unsigned short*)Cout)[(long)mg * N + ng] = f32_to_bf16(val);
      else
        ((float*)Cout)[(long)mg * N + ng] = val;
    }
  }
}

// ---------------- router FC2 + softmax + top-2 + renorm ----------------
__global__ void router_fc2_top2(const unsigned short* __restrict__ hb,
                                const float* __restrict__ Wr2,
                                const float* __restrict__ br2,
                                int* __restrict__ tok_e,
                                float* __restrict__ tok_p,
                                int* __restrict__ cnt)
{
  int t = blockIdx.x * blockDim.x + threadIdx.x;
  if (t >= T_TOK) return;
  float logit[EE];
  #pragma unroll
  for (int e = 0; e < EE; ++e) logit[e] = br2[e];
  const unsigned short* hrow = hb + (long)t * DD;
  for (int d = 0; d < DD; ++d) {
    float hv = bf16_to_f32(hrow[d]);
    const float* w = Wr2 + (long)d * EE;
    #pragma unroll
    for (int e = 0; e < EE; ++e) logit[e] += hv * w[e];
  }
  float mx = logit[0];
  #pragma unroll
  for (int e = 1; e < EE; ++e) mx = fmaxf(mx, logit[e]);
  float p[EE];
  #pragma unroll
  for (int e = 0; e < EE; ++e) p[e] = __expf(logit[e] - mx);
  int i0 = 0;
  #pragma unroll
  for (int e = 1; e < EE; ++e) if (p[e] > p[i0]) i0 = e;
  int i1 = (i0 == 0) ? 1 : 0;
  #pragma unroll
  for (int e = 0; e < EE; ++e) if (e != i0 && p[e] > p[i1]) i1 = e;
  float v0 = p[i0], v1 = p[i1], s = v0 + v1;
  tok_e[t * 2 + 0] = i0;  tok_e[t * 2 + 1] = i1;
  tok_p[t * 2 + 0] = v0 / s;  tok_p[t * 2 + 1] = v1 / s;
  atomicAdd(&cnt[i0], 1);
  atomicAdd(&cnt[i1], 1);
}

// ---------------- segment scan (128-aligned offsets) ----------------
__global__ void scan_offsets(const int* __restrict__ cnt,
                             int* __restrict__ seg_off,
                             int* __restrict__ cursor) {
  if (threadIdx.x == 0 && blockIdx.x == 0) {
    int off = 0;
    for (int e = 0; e < EE; ++e) {
      seg_off[e] = off;
      off += (cnt[e] + 127) & ~127;
      cursor[e] = 0;
    }
    seg_off[EE] = off;
  }
}

// ---------------- build per-expert token lists + slot table ----------------
__global__ void build_lists(const int* __restrict__ tok_e,
                            const int* __restrict__ seg_off,
                            int* __restrict__ cursor,
                            int* __restrict__ rowtok,
                            int* __restrict__ tok_slot)
{
  int t = blockIdx.x * blockDim.x + threadIdx.x;
  if (t >= T_TOK) return;
  #pragma unroll
  for (int k = 0; k < KSEL; ++k) {
    int e = tok_e[t * 2 + k];
    int pos = atomicAdd(&cursor[e], 1);
    int slot = seg_off[e] + pos;
    rowtok[slot] = t;
    tok_slot[t * 2 + k] = slot;
  }
}

// ---------------- combine: out[t] = p0*y[slot0] + p1*y[slot1] ----------------
__global__ void combine_kernel(const float* __restrict__ yseg,
                               const int* __restrict__ tok_slot,
                               const float* __restrict__ tok_p,
                               float* __restrict__ out)
{
  int i = blockIdx.x * blockDim.x + threadIdx.x;   // over T*D/4
  if (i >= T_TOK * DD / 4) return;
  int t = i / (DD / 4);
  int j = i % (DD / 4);
  int   s0 = tok_slot[t * 2 + 0], s1 = tok_slot[t * 2 + 1];
  float p0 = tok_p[t * 2 + 0],    p1 = tok_p[t * 2 + 1];
  float4 a = ((const float4*)(yseg + (long)s0 * DD))[j];
  float4 b = ((const float4*)(yseg + (long)s1 * DD))[j];
  float4 r;
  r.x = p0 * a.x + p1 * b.x;
  r.y = p0 * a.y + p1 * b.y;
  r.z = p0 * a.z + p1 * b.z;
  r.w = p0 * a.w + p1 * b.w;
  ((float4*)out)[i] = r;
}

// ---------------- host side ----------------
static inline size_t align256(size_t x) { return (x + 255) & ~(size_t)255; }

extern "C" void kernel_launch(void* const* d_in, const int* in_sizes, int n_in,
                              void* d_out, int out_size, void* d_ws, size_t ws_size,
                              hipStream_t stream) {
  (void)in_sizes; (void)n_in; (void)out_size; (void)ws_size;
  const float* x   = (const float*)d_in[0];
  const float* Wr1 = (const float*)d_in[1];
  const float* br1 = (const float*)d_in[2];
  const float* Wr2 = (const float*)d_in[3];
  const float* br2 = (const float*)d_in[4];
  const float* W1  = (const float*)d_in[5];
  const float* b1  = (const float*)d_in[6];
  const float* W2  = (const float*)d_in[7];
  const float* b2  = (const float*)d_in[8];
  float* out = (float*)d_out;

  // workspace carve-up (~385 MB)
  char* ws = (char*)d_ws;
  size_t off = 0;
  unsigned short* xb   = (unsigned short*)(ws + off); off = align256(off + (size_t)T_TOK * DD * 2);
  unsigned short* wr1b = (unsigned short*)(ws + off); off = align256(off + (size_t)DD * DD * 2);
  unsigned short* hb   = (unsigned short*)(ws + off); off = align256(off + (size_t)T_TOK * DD * 2);
  unsigned short* w1b  = (unsigned short*)(ws + off); off = align256(off + (size_t)EE * DD * HH * 2);
  unsigned short* w2b  = (unsigned short*)(ws + off); off = align256(off + (size_t)EE * HH * DD * 2);
  unsigned short* hseg = (unsigned short*)(ws + off); off = align256(off + (size_t)PMAX * HH * 2);
  float* yseg          = (float*)(ws + off);          off = align256(off + (size_t)PMAX * DD * 4);
  int*   tok_e         = (int*)(ws + off);            off = align256(off + (size_t)T_TOK * 2 * 4);
  float* tok_p         = (float*)(ws + off);          off = align256(off + (size_t)T_TOK * 2 * 4);
  int*   tok_slot      = (int*)(ws + off);            off = align256(off + (size_t)T_TOK * 2 * 4);
  int*   rowtok        = (int*)(ws + off);            off = align256(off + (size_t)PMAX * 4);
  int*   cnt           = (int*)(ws + off);            off = align256(off + 16 * 4);
  int*   cursor        = (int*)(ws + off);            off = align256(off + 16 * 4);
  int*   seg_off       = (int*)(ws + off);            off = align256(off + 16 * 4);

  const int TB = 256;
  auto blocks = [](long n, int tb) { return (unsigned)((n + tb - 1) / tb); };

  // 1) convert inputs / weights to bf16
  cvt_bf16_kernel<<<blocks((long)T_TOK * DD, TB), TB, 0, stream>>>(x, xb, (long)T_TOK * DD);
  cvt_bf16_kernel<<<blocks((long)DD * DD, TB), TB, 0, stream>>>(Wr1, wr1b, (long)DD * DD);
  cvt_bf16_kernel<<<blocks((long)EE * DD * HH, TB), TB, 0, stream>>>(W1, w1b, (long)EE * DD * HH);
  cvt_bf16_kernel<<<blocks((long)EE * HH * DD, TB), TB, 0, stream>>>(W2, w2b, (long)EE * HH * DD);
  fill_int_kernel<<<1, 64, 0, stream>>>(cnt, 0, EE);

  // 2) router FC1: hb = gelu(x @ Wr1 + br1)   [8192 x 1024 x 1024]
  gemm_bf16_wmma<true, true><<<dim3(DD / 128, T_TOK / 128), 256, 0, stream>>>(
      xb, nullptr, wr1b, nullptr, 0, br1, 0, hb, DD, DD);

  // 3) router FC2 + softmax + top-2 + renorm
  router_fc2_top2<<<blocks(T_TOK, TB), TB, 0, stream>>>(hb, Wr2, br2, tok_e, tok_p, cnt);

  // 4) segment offsets, token lists
  scan_offsets<<<1, 1, 0, stream>>>(cnt, seg_off, cursor);
  fill_int_kernel<<<blocks(PMAX, TB), TB, 0, stream>>>(rowtok, -1, PMAX);
  build_lists<<<blocks(T_TOK, TB), TB, 0, stream>>>(tok_e, seg_off, cursor, rowtok, tok_slot);

  // 5) expert FC1 (gathered rows): hseg = gelu(x[rowtok] @ W1[e] + b1[e])
  gemm_bf16_wmma<true, true><<<dim3(HH / 128, PMAX / 128), 256, 0, stream>>>(
      xb, rowtok, w1b, seg_off, (long)DD * HH, b1, (long)HH, hseg, HH, DD);

  // 6) expert FC2: yseg = hseg @ W2[e] + b2[e]   (f32 out)
  gemm_bf16_wmma<false, false><<<dim3(DD / 128, PMAX / 128), 256, 0, stream>>>(
      hseg, nullptr, w2b, seg_off, (long)HH * DD, b2, (long)DD, yseg, DD, HH);

  // 7) combine with renormalized top-2 probs
  combine_kernel<<<blocks((long)T_TOK * DD / 4, TB), TB, 0, stream>>>(
      yseg, tok_slot, tok_p, out);
}
